// TFNO_9818295238750
// MI455X (gfx1250) — matrix operand order repaired and checked by reading, not compiled, not run on hardware
//
#include <hip/hip_runtime.h>
#include <math.h>

// ---------------------------------------------------------------------------
// TFNO forward on gfx1250 (MI455X).
// All matmuls (lift/proj MLPs, truncated DFTs, complex mode-mix, skip) are
// wave32 WMMA f16->f32 GEMMs. Every WMMA B-operand lives in a column-major
// f16 layout so each lane loads two 8-half K runs (2x global_load_b128).
// gelu = x*sigmoid(2u): one v_exp_f32 + one v_rcp_f32, branch-free.
// 8 waves / 256-thread block.
// ---------------------------------------------------------------------------

typedef __attribute__((ext_vector_type(16))) _Float16 v16h;
typedef __attribute__((ext_vector_type(8)))  _Float16 v8h;
typedef __attribute__((ext_vector_type(8)))  float    v8f;

#define BB   8
#define HID  64
#define NPIX (BB*256*256)   // 524288 pixels

__device__ __forceinline__ v8f wmma32(v16h a, v16h b, v8f c){
  return __builtin_amdgcn_wmma_f32_16x16x32_f16(false, a, false, b, (short)0, c, false, false);
}

// tanh-gelu via sigmoid identity: 0.5x(1+tanh(u)) == x*sigmoid(2u).
// v_exp_f32 + v_rcp_f32, no IEEE-div expansion, no branches.
__device__ __forceinline__ float gelu_tanh(float x){
  float t2u = 1.5957691216057308f*(x + 0.044715f*x*x*x);   // 2*sqrt(2/pi)*(...)
  float e = __expf(-t2u);
  return x * __builtin_amdgcn_rcpf(1.0f + e);
}

// A-tile (16x32 f16) from a row-major f16 row (contiguous K): 2x16B loads.
__device__ __forceinline__ v16h load_a16(const _Float16* rowp, int k0, int lane){
  int kb = (lane >= 16) ? 8 : 0;
  v8h lo = *(const v8h*)(rowp + k0 + kb);
  v8h hi = *(const v8h*)(rowp + k0 + kb + 16);
  v16h a;
#pragma unroll
  for (int i = 0; i < 8; ++i){ a[i] = lo[i]; a[8+i] = hi[i]; }
  return a;
}
// A-tile from f32 row (contiguous K), converted to f16.
__device__ __forceinline__ v16h load_a_f32(const float* rowp, int k0, int lane){
  int kb = (lane >= 16) ? 8 : 0;
  v16h a;
#pragma unroll
  for (int i = 0; i < 8; ++i){
    a[i]   = (_Float16)rowp[k0 + kb + i];
    a[8+i] = (_Float16)rowp[k0 + kb + 16 + i];
  }
  return a;
}
// B-tile (32x16 f16) from column-major storage: colp points at this lane's
// column (length-K contiguous); two 8-half runs (k0+kb, k0+kb+16) -> 2x global_load_b128.
__device__ __forceinline__ v16h load_b16(const _Float16* colp, int k0, int lane){
  int kb2 = (lane >= 16) ? 8 : 0;
  v8h x0 = *(const v8h*)(colp + k0 + kb2);
  v8h x1 = *(const v8h*)(colp + k0 + kb2 + 16);
  v16h b;
#pragma unroll
  for (int i = 0; i < 8; ++i){ b[i] = x0[i]; b[8+i] = x1[i]; }
  return b;
}

// --------------------------- init kernels ----------------------------------

__global__ void k_tables(_Float16* TxF, _Float16* TyF, _Float16* TyIr,
                         _Float16* TyIi, _Float16* T5){
  int i = blockIdx.x*blockDim.x + threadIdx.x;
  const float TP = 6.2831853071795864f / 256.0f;
  if (i < 32*256){                      // forward x: rows<16 cos, rows>=16 -sin
    int r = i >> 8, x = i & 255;
    int ph = ((r & 15) * x) & 255;
    float th = TP*(float)ph;
    TxF[i] = (_Float16)((r < 16) ? cosf(th) : -sinf(th));
  }
  if (i < 64*256){                      // forward y: rows<32 cos, rows>=32 sin
    int r = i >> 8, y = i & 255;
    int mm = r & 31;
    int ky = (mm < 16) ? mm : (mm + 224);
    int ph = (ky * y) & 255;
    float th = TP*(float)ph;
    TyF[i] = (_Float16)((r < 32) ? cosf(th) : sinf(th));
  }
  if (i < 256*64){                      // inverse y (e^{+i}) real/imag tables
    int y = i >> 6, k = i & 63;
    int mm = k & 31;
    int ky = (mm < 16) ? mm : (mm + 224);
    int ph = (ky * y) & 255;
    float c = cosf(TP*(float)ph), s = sinf(TP*(float)ph);
    TyIr[i] = (_Float16)((k < 32) ? c : -s);   // Zr = cos*Gr - sin*Gi
    TyIi[i] = (_Float16)((k < 32) ? s : c);    // Zi = sin*Gr + cos*Gi
  }
  if (i < 256*32){                      // inverse x real (Hermitian weights)
    int x = i >> 5, k2 = i & 31;
    int kx = k2 & 15;
    float w = (kx == 0) ? 1.0f : 2.0f;
    int ph = (kx * x) & 255;
    float th = TP*(float)ph;
    T5[i] = (_Float16)((k2 < 16) ? w*cosf(th) : -w*sinf(th));
  }
}

// Column-major (B-ready) f16 weights: WT[n][k].
__global__ void k_cvt(const float* lw1, const float* lw2, const float* skw,
                      const float* pw1, const float* pw2,
                      _Float16* w1T, _Float16* w2T, _Float16* skT,
                      _Float16* pw1T, _Float16* pw2T){
  int i = blockIdx.x*blockDim.x + threadIdx.x;
  if (i < 256*32){ int n = i >> 5, k = i & 31;               // lift1: K 20->32 pad
    w1T[i] = (k < 20) ? (_Float16)lw1[k*256+n] : (_Float16)0.f; }
  if (i < 64*256){ int n = i >> 8, k = i & 255;              // lift2
    w2T[i] = (_Float16)lw2[k*64+n]; }
  if (i < 4*64*64){ int l = i >> 12, n = (i >> 6) & 63, k = i & 63;  // skip
    skT[i] = (_Float16)skw[(l*64+k)*64+n]; }
  if (i < 256*64){ int n = i >> 6, k = i & 63;               // proj1
    pw1T[i] = (_Float16)pw1[k*256+n]; }
  if (i < 16*256){ int n = i >> 8, k = i & 255;              // proj2: N 5->16 pad
    pw2T[i] = (n < 5) ? (_Float16)pw2[k*5+n] : (_Float16)0.f; }
}

// Spectral weights -> f16 [l][mode(mi,kx)][o][i] (B-ready per mode).
__global__ void k_cvt_spec(const float* w1r, const float* w1i,
                           const float* w2r, const float* w2i,
                           _Float16* WrT, _Float16* WiT){
  int idx = blockIdx.x*blockDim.x + threadIdx.x;   // 4*512*64*64 = 8388608
  int ii   = idx & 63;
  int o    = (idx >> 6) & 63;
  int mode = (idx >> 12) & 511;
  int l    = idx >> 21;
  int mi = mode >> 4, kx = mode & 15;
  bool top = (mi < 16);
  int g = top ? mi : (mi - 16);
  size_t so = (size_t)l*1048576 + (size_t)ii*16384 + (size_t)o*256 + g*16 + kx;
  WrT[idx] = (_Float16)(top ? w1r[so] : w2r[so]);
  WiT[idx] = (_Float16)(top ? w1i[so] : w2i[so]);
}

// --------------------------- network kernels -------------------------------

// Lift: t = gelu(X(16x20)@W1+b1) in LDS; h = t@W2+b2 -> bufA (f32) + hT (f16 T).
__global__ void k_lift(const float* __restrict__ x, const _Float16* __restrict__ w1T,
                       const float* __restrict__ b1, const _Float16* __restrict__ w2T,
                       const float* __restrict__ b2, float* __restrict__ hout,
                       _Float16* __restrict__ hT){
  __shared__ _Float16 t[8][16][256];
  int w = threadIdx.x >> 5, lane = threadIdx.x & 31;
  int rt = blockIdx.x*8 + w;
  size_t p0 = (size_t)rt * 16;
  int row = lane & 15, lcol = lane & 15;
  int rbase = (lane >= 16) ? 8 : 0;
  size_t p = p0 + row;
  int bidx = (int)(p >> 16), yx = (int)(p & 65535);
  int kb = (lane >= 16) ? 8 : 0;
  v16h a;
#pragma unroll
  for (int i = 0; i < 8; ++i){
    int k0 = kb + i;                      // <=15 < 20 always valid
    a[i] = (_Float16)x[((size_t)bidx*20 + k0)*65536 + yx];
    int k1 = kb + 16 + i;                 // 16..31 guard K=20
    a[8+i] = (k1 < 20) ? (_Float16)x[((size_t)bidx*20 + k1)*65536 + yx]
                       : (_Float16)0.f;
  }
  for (int nt = 0; nt < 16; ++nt){        // N=256
    int col = nt*16 + lcol;
    v8f acc = {};
    acc = wmma32(a, load_b16(w1T + (size_t)col*32, 0, lane), acc);
    float bias = b1[col];
#pragma unroll
    for (int i = 0; i < 8; ++i) t[w][rbase+i][col] = (_Float16)gelu_tanh(acc[i] + bias);
  }
  __syncthreads();
  int by = (int)(p0 >> 8);                // b*256+y (16 | 256)
  int x0 = (int)(p0 & 255);
  for (int nt = 0; nt < 4; ++nt){         // N=64, K=256
    int col = nt*16 + lcol;
    v8f acc = {};
    for (int kc = 0; kc < 8; ++kc)
      acc = wmma32(load_a16(&t[w][row][0], kc*32, lane),
                   load_b16(w2T + (size_t)col*256, kc*32, lane), acc);
    float bias = b2[col];
    v8h hv;
#pragma unroll
    for (int i = 0; i < 8; ++i){
      float v = acc[i] + bias;
      hout[(p0 + rbase + i)*64 + col] = v;
      hv[i] = (_Float16)v;
    }
    *(v8h*)(hT + ((size_t)by*64 + col)*256 + x0 + rbase) = hv;   // transposed copy
  }
}

// Forward x-DFT: per (b,y): Fx(32x64) = TxF(32x256) @ hT_row(256x64 col-major).
__global__ void k_xdft(const _Float16* __restrict__ hT, const _Float16* __restrict__ TxF,
                       _Float16* __restrict__ FxT){
  int w = threadIdx.x >> 5, lane = threadIdx.x & 31;
  int tid = blockIdx.x*8 + w;             // 16384
  int by = tid >> 3, nt = (tid >> 1) & 3, mt = tid & 1;
  int arow = mt*16 + (lane & 15);
  int bcol = nt*16 + (lane & 15);
  const _Float16* colp = hT + ((size_t)by*64 + bcol)*256;
  v8f acc = {};
  for (int kc = 0; kc < 8; ++kc){
    if (kc < 7) __builtin_prefetch(colp + (kc+1)*32, 0, 0);
    acc = wmma32(load_a16(TxF + arow*256, kc*32, lane),
                 load_b16(colp, kc*32, lane), acc);
  }
  int b = by >> 8, y = by & 255;
  int rbase = mt*16 + ((lane >= 16) ? 8 : 0);
#pragma unroll
  for (int i = 0; i < 8; ++i)             // FxT[b][k2*64+ch][y]
    FxT[((size_t)b*2048 + (size_t)(rbase+i)*64 + bcol)*256 + y] = (_Float16)acc[i];
}

// Forward y-DFT: per b: U(64x2048) = TyF(64x256) @ FxT_b(256x2048 col-major).
__global__ void k_ydft(const _Float16* __restrict__ FxT, const _Float16* __restrict__ TyF,
                       float* __restrict__ U){
  int w = threadIdx.x >> 5, lane = threadIdx.x & 31;
  int tid = blockIdx.x*8 + w;             // 4096
  int b = tid >> 9, nt = (tid >> 2) & 127, mt = tid & 3;
  int arow = mt*16 + (lane & 15);
  int bcol = nt*16 + (lane & 15);
  const _Float16* colp = FxT + ((size_t)b*2048 + bcol)*256;
  v8f acc = {};
  for (int kc = 0; kc < 8; ++kc){
    if (kc < 7) __builtin_prefetch(colp + (kc+1)*32, 0, 0);
    acc = wmma32(load_a16(TyF + arow*256, kc*32, lane),
                 load_b16(colp, kc*32, lane), acc);
  }
  float* Ub = U + (size_t)b * 64 * 2048;
  int rbase = mt*16 + ((lane >= 16) ? 8 : 0);
#pragma unroll
  for (int i = 0; i < 8; ++i) Ub[(size_t)(rbase + i)*2048 + bcol] = acc[i];
}

// Complex combine: Fyr = C@r + S@i ; Fyi = C@i - S@r.
__global__ void k_pack(const float* __restrict__ U, float* __restrict__ Fyr,
                       float* __restrict__ Fyi){
  int idx = blockIdx.x*blockDim.x + threadIdx.x;      // 262144
  int ch = idx & 63, kx = (idx >> 6) & 15, m = (idx >> 10) & 31, b = idx >> 15;
  const float* Ub = U + (size_t)b * 64 * 2048;
  float cr = Ub[m*2048 + kx*64 + ch];
  float ci = Ub[m*2048 + (16+kx)*64 + ch];
  float sr = Ub[(32+m)*2048 + kx*64 + ch];
  float si = Ub[(32+m)*2048 + (16+kx)*64 + ch];
  Fyr[idx] = cr + si;
  Fyi[idx] = ci - sr;
}

// Mode-wise complex 64x64 mix, B=8 batched in one tile (rows 0..7 Re, 8..15 Im),
// cross terms combined via shfl_xor(16). Output GT[b][n=(kx,ch)][k] f16.
__global__ void k_modes(const float* __restrict__ Fyr, const float* __restrict__ Fyi,
                        const _Float16* __restrict__ WrT, const _Float16* __restrict__ WiT,
                        _Float16* __restrict__ GT){
  int w = threadIdx.x >> 5, lane = threadIdx.x & 31;
  int tid = blockIdx.x*8 + w;             // 2048
  int mode = tid >> 2, nt = tid & 3;
  int mi = mode >> 4, kx = mode & 15;
  int row = lane & 15;
  int bnum = row & 7;
  const float* arow = ((row < 8) ? Fyr : Fyi) + (((size_t)bnum*32 + mi)*16 + kx)*64;
  int col = nt*16 + (lane & 15);
  const _Float16* wrc = WrT + (size_t)mode*4096 + (size_t)col*64;
  const _Float16* wic = WiT + (size_t)mode*4096 + (size_t)col*64;
  v8f acc1 = {}, acc2 = {};
  for (int kc = 0; kc < 2; ++kc){
    v16h a = load_a_f32(arow, kc*32, lane);
    acc1 = wmma32(a, load_b16(wrc, kc*32, lane), acc1);  // [Fr;Fi]@Wr
    acc2 = wmma32(a, load_b16(wic, kc*32, lane), acc2);  // [Fr;Fi]@Wi
  }
#pragma unroll
  for (int i = 0; i < 8; ++i){
    float o1 = __shfl_xor(acc1[i], 16, 32);   // Fi@Wr (other half rows)
    float o2 = __shfl_xor(acc2[i], 16, 32);   // Fi@Wi
    if (lane < 16){
      float Gr = acc1[i] - o2;
      float Gi = acc2[i] + o1;
      size_t nb = ((size_t)i*1024 + (size_t)kx*64 + col)*64;
      GT[nb + mi]      = (_Float16)Gr;
      GT[nb + 32 + mi] = (_Float16)Gi;
    }
  }
}

// Inverse y-DFT: per (b,sel): Z(256x1024) = TyI(256x64) @ GT_b(64x1024 col-major).
// Applies 1/(H*W); writes ZT[b,y][ch][k2] f16 (B-ready for inverse-x).
__global__ void k_iydft(const _Float16* __restrict__ GT, const _Float16* __restrict__ TyIr,
                        const _Float16* __restrict__ TyIi, _Float16* __restrict__ ZT){
  int w = threadIdx.x >> 5, lane = threadIdx.x & 31;
  int tid = blockIdx.x*8 + w;             // 16384
  int bsel = tid >> 10, nt = (tid >> 4) & 63, mt = tid & 15;
  int b = bsel >> 1, sel = bsel & 1;
  const _Float16* T = sel ? TyIi : TyIr;
  int arow = mt*16 + (lane & 15);
  int bcol = nt*16 + (lane & 15);
  const _Float16* colp = GT + ((size_t)b*1024 + bcol)*64;
  v8f acc = {};
  for (int kc = 0; kc < 2; ++kc)
    acc = wmma32(load_a16(T + arow*64, kc*32, lane),
                 load_b16(colp, kc*32, lane), acc);
  int kx = bcol >> 6, ch = bcol & 63;
  int k2 = sel*16 + kx;
  int rbase = mt*16 + ((lane >= 16) ? 8 : 0);
  const float SC = 1.0f/65536.0f;
#pragma unroll
  for (int i = 0; i < 8; ++i)
    ZT[(((size_t)b*256 + (rbase+i))*64 + ch)*32 + k2] = (_Float16)(acc[i]*SC);
}

// Inverse x (real): per (b,y): sp(256x64) = T5(256x32) @ ZT_row(32x64 col-major).
__global__ void k_ixdft(const _Float16* __restrict__ ZT, const _Float16* __restrict__ T5,
                        float* __restrict__ sp){
  int w = threadIdx.x >> 5, lane = threadIdx.x & 31;
  int tid = blockIdx.x*8 + w;             // 131072
  int by = tid >> 6, nt = (tid >> 4) & 3, mt = tid & 15;
  int arow = mt*16 + (lane & 15);
  int bcol = nt*16 + (lane & 15);
  const _Float16* colp = ZT + (size_t)by*2048 + (size_t)bcol*32;
  v8f acc = {};
  acc = wmma32(load_a16(T5 + arow*32, 0, lane), load_b16(colp, 0, lane), acc);
  float* out = sp + (size_t)by * 256 * 64;
  int rbase = mt*16 + ((lane >= 16) ? 8 : 0);
#pragma unroll
  for (int i = 0; i < 8; ++i) out[(size_t)(rbase + i)*64 + bcol] = acc[i];
}

// Skip: spb = (gelu?)(spb + h@skip_w + b); writes spb (f32) + hT (f16 transposed).
__global__ void k_skip(const float* __restrict__ h, float* __restrict__ spb,
                       const _Float16* __restrict__ skT, const float* __restrict__ skb,
                       int dogelu, _Float16* __restrict__ hT){
  int w = threadIdx.x >> 5, lane = threadIdx.x & 31;
  int tid = blockIdx.x*8 + w;             // 131072
  int rt = tid >> 2, nt = tid & 3;
  size_t p0 = (size_t)rt * 16;
  int row = lane & 15;
  int col = nt*16 + (lane & 15);
  int rbase = (lane >= 16) ? 8 : 0;
  const float* arow = h + (p0 + row)*64;
  float bias = skb[col];
  v8f acc;
#pragma unroll
  for (int i = 0; i < 8; ++i) acc[i] = spb[(p0 + rbase + i)*64 + col] + bias;
  const _Float16* wc = skT + (size_t)col*64;
  for (int kc = 0; kc < 2; ++kc)
    acc = wmma32(load_a_f32(arow, kc*32, lane), load_b16(wc, kc*32, lane), acc);
  int by = (int)(p0 >> 8);
  int x0 = (int)(p0 & 255);
  v8h hv;
#pragma unroll
  for (int i = 0; i < 8; ++i){
    float v = acc[i];
    if (dogelu) v = gelu_tanh(v);
    spb[(p0 + rbase + i)*64 + col] = v;
    hv[i] = (_Float16)v;
  }
  *(v8h*)(hT + ((size_t)by*64 + col)*256 + x0 + rbase) = hv;
}

// Projection: t = gelu(h(16x64)@W1+b1) in LDS; out = t@W2T+b2 (N padded 5->16).
__global__ void k_proj(const float* __restrict__ h, const _Float16* __restrict__ w1T,
                       const float* __restrict__ b1, const _Float16* __restrict__ w2T,
                       const float* __restrict__ b2, float* __restrict__ out){
  __shared__ _Float16 t[8][16][256];
  int w = threadIdx.x >> 5, lane = threadIdx.x & 31;
  int rt = blockIdx.x*8 + w;
  size_t p0 = (size_t)rt * 16;
  int row = lane & 15, lcol = lane & 15;
  int rbase = (lane >= 16) ? 8 : 0;
  const float* arow = h + (p0 + row)*64;
  for (int nt = 0; nt < 16; ++nt){
    int col = nt*16 + lcol;
    v8f acc = {};
    for (int kc = 0; kc < 2; ++kc)
      acc = wmma32(load_a_f32(arow, kc*32, lane),
                   load_b16(w1T + (size_t)col*64, kc*32, lane), acc);
    float bias = b1[col];
#pragma unroll
    for (int i = 0; i < 8; ++i) t[w][rbase+i][col] = (_Float16)gelu_tanh(acc[i] + bias);
  }
  __syncthreads();
  v8f acc = {};
  for (int kc = 0; kc < 8; ++kc)
    acc = wmma32(load_a16(&t[w][row][0], kc*32, lane),
                 load_b16(w2T + (size_t)lcol*256, kc*32, lane), acc);
  if (lcol < 5){
    float bias = b2[lcol];
#pragma unroll
    for (int i = 0; i < 8; ++i){
      size_t p = p0 + rbase + i;
      int bidx = (int)(p >> 16), yx = (int)(p & 65535);
      out[((size_t)bidx*5 + lcol)*65536 + yx] = acc[i] + bias;
    }
  }
}

// --------------------------- host launcher ---------------------------------

extern "C" void kernel_launch(void* const* d_in, const int* in_sizes, int n_in,
                              void* d_out, int out_size, void* d_ws, size_t ws_size,
                              hipStream_t stream){
  (void)in_sizes; (void)n_in; (void)out_size; (void)ws_size;
  const float* x        = (const float*)d_in[0];
  const float* lift_w1  = (const float*)d_in[1];
  const float* lift_b1  = (const float*)d_in[2];
  const float* lift_w2  = (const float*)d_in[3];
  const float* lift_b2  = (const float*)d_in[4];
  const float* spec_w1r = (const float*)d_in[5];
  const float* spec_w1i = (const float*)d_in[6];
  const float* spec_w2r = (const float*)d_in[7];
  const float* spec_w2i = (const float*)d_in[8];
  const float* skip_w   = (const float*)d_in[9];
  const float* skip_b   = (const float*)d_in[10];
  const float* proj_w1  = (const float*)d_in[11];
  const float* proj_b1  = (const float*)d_in[12];
  const float* proj_w2  = (const float*)d_in[13];
  const float* proj_b2  = (const float*)d_in[14];
  float* out = (float*)d_out;

  size_t off = 0;
  auto carve = [&](size_t bytes)->char*{
    char* p = (char*)d_ws + off;
    off += (bytes + 255) & ~(size_t)255;
    return p;
  };
  float*    bufA = (float*)carve((size_t)NPIX*64*4);            // f32 act ping
  float*    bufB = (float*)carve((size_t)NPIX*64*4);            // f32 act pong / sp
  _Float16* hT   = (_Float16*)carve((size_t)2048*64*256*2);     // h transposed f16
  _Float16* FxT  = (_Float16*)carve((size_t)BB*2048*256*2);     // x-DFT out (col-major)
  float*    U    = (float*)carve((size_t)BB*64*2048*4);         // y-DFT raw f32
  float*    Fyr  = (float*)carve((size_t)BB*32*16*64*4);
  float*    Fyi  = (float*)carve((size_t)BB*32*16*64*4);
  _Float16* GT   = (_Float16*)carve((size_t)BB*1024*64*2);      // mode out (col-major)
  _Float16* ZT   = (_Float16*)carve((size_t)BB*256*64*32*2);    // inv-y out (col-major)
  _Float16* WrT  = (_Float16*)carve((size_t)4*512*4096*2);      // spec weights f16 T
  _Float16* WiT  = (_Float16*)carve((size_t)4*512*4096*2);
  _Float16* w1T  = (_Float16*)carve(256*32*2);
  _Float16* w2T  = (_Float16*)carve(64*256*2);
  _Float16* skT  = (_Float16*)carve(4*64*64*2);
  _Float16* pw1T = (_Float16*)carve(256*64*2);
  _Float16* pw2T = (_Float16*)carve(16*256*2);
  _Float16* TxF  = (_Float16*)carve(32*256*2);
  _Float16* TyF  = (_Float16*)carve(64*256*2);
  _Float16* TyIr = (_Float16*)carve(256*64*2);
  _Float16* TyIi = (_Float16*)carve(256*64*2);
  _Float16* T5   = (_Float16*)carve(256*32*2);

  k_tables<<<64, 256, 0, stream>>>(TxF, TyF, TyIr, TyIi, T5);
  k_cvt<<<64, 256, 0, stream>>>(lift_w1, lift_w2, skip_w, proj_w1, proj_w2,
                                w1T, w2T, skT, pw1T, pw2T);
  k_cvt_spec<<<32768, 256, 0, stream>>>(spec_w1r, spec_w1i, spec_w2r, spec_w2i,
                                        WrT, WiT);
  k_lift<<<4096, 256, 0, stream>>>(x, w1T, lift_b1, w2T, lift_b2, bufA, hT);

  float* hcur = bufA;
  float* hsp  = bufB;
  for (int l = 0; l < 4; ++l){
    size_t wl = (size_t)l * 512 * 4096;
    k_xdft <<<2048,  256, 0, stream>>>(hT, TxF, FxT);
    k_ydft <<<512,   256, 0, stream>>>(FxT, TyF, U);
    k_pack <<<1024,  256, 0, stream>>>(U, Fyr, Fyi);
    k_modes<<<256,   256, 0, stream>>>(Fyr, Fyi, WrT + wl, WiT + wl, GT);
    k_iydft<<<2048,  256, 0, stream>>>(GT, TyIr, TyIi, ZT);
    k_ixdft<<<16384, 256, 0, stream>>>(ZT, T5, hsp);
    k_skip <<<16384, 256, 0, stream>>>(hcur, hsp, skT + (size_t)l*4096,
                                       skip_b + l*64, (l < 3) ? 1 : 0, hT);
    float* tswap = hcur; hcur = hsp; hsp = tswap;   // A->B->A->B->A
  }
  k_proj<<<4096, 256, 0, stream>>>(hcur, pw1T, proj_b1, pw2T, proj_b2, out);
}
